// HybridLocalPoolPointnet_90769838834283
// MI455X (gfx1250) — matrix operand order repaired and checked by reading, not compiled, not run on hardware
//
#include <hip/hip_runtime.h>

typedef __attribute__((ext_vector_type(16))) __bf16 v16bf;
typedef __attribute__((ext_vector_type(8)))  float  v8f;

#define NB   8
#define TPT  30000
#define BT   240000
#define HID  128
#define LDA  256
#define R2   16384
#define NT   235   // ceil(30000/128)

union AFrag { v16bf v; unsigned short u[16]; uint4 q[2]; };

__device__ __forceinline__ unsigned short f2bf(float f) {
  unsigned int u = __float_as_uint(f);
  unsigned int r = u + 0x7FFFu + ((u >> 16) & 1u);
  return (unsigned short)(r >> 16);
}
__device__ __forceinline__ float bf2f(unsigned short h) {
  return __uint_as_float(((unsigned int)h) << 16);
}
__device__ __forceinline__ unsigned encf(float f) {
  unsigned u = __float_as_uint(f);
  return (u & 0x80000000u) ? ~u : (u | 0x80000000u);
}
__device__ __forceinline__ float decf(unsigned e) {
  unsigned u = (e & 0x80000000u) ? (e ^ 0x80000000u) : ~e;
  return __uint_as_float(u);
}

__device__ __forceinline__ v8f wmma_bf16(const AFrag& a, const AFrag& b, v8f c) {
  return __builtin_amdgcn_wmma_f32_16x16x32_bf16(false, a.v, false, b.v, (short)0, c, false, false);
}

// A fragment from row-major bf16 rows (LDS or global), rows contiguous, no clamp.
__device__ __forceinline__ AFrag afrag_rows(const unsigned short* row0, int ld, int k0, int lane) {
  const unsigned short* pr = row0 + (lane & 15) * ld + k0 + ((lane & 16) ? 8 : 0);
  AFrag a;
  a.q[0] = *(const uint4*)pr;
  a.q[1] = *(const uint4*)(pr + 16);
  return a;
}

// A fragment from global activation buffer with per-lane row clamp, optional relu.
__device__ __forceinline__ AFrag afrag_gclamp(const unsigned short* abase, int ld, int k0,
                                              int t0, int tmax, int lane, int relu) {
  int t = t0 + (lane & 15); if (t > tmax) t = tmax;
  const unsigned short* pr = abase + (size_t)t * ld + k0 + ((lane & 16) ? 8 : 0);
  AFrag a;
  a.q[0] = *(const uint4*)pr;
  a.q[1] = *(const uint4*)(pr + 16);
  if (relu) {
#pragma unroll
    for (int i = 0; i < 16; ++i) if (a.u[i] & 0x8000u) a.u[i] = 0;
  }
  return a;
}

// B fragment: lane <-> N (lane&15), elements over K in table order (k = 8h+s, 16+8h+s-8).
// w points at bf16 weight [K][128] row-major.
__device__ __forceinline__ AFrag bfrag(const unsigned short* w, int kt, int nc, int lane) {
  const unsigned short* p = w + ((kt << 5) << 7) + (nc << 4) + (lane & 15);
  int h8 = (lane & 16) ? 8 : 0;
  AFrag b;
#pragma unroll
  for (int s = 0; s < 16; ++s) {
    int k = (s < 8) ? (h8 + s) : (16 + h8 + s - 8);
    b.u[s] = p[k << 7];
  }
  return b;
}

// ---------------------------------------------------------------- utilities
__global__ void k_fill(unsigned* p, unsigned v, unsigned n) {
  unsigned n4 = n >> 2;
  uint4 vv; vv.x = v; vv.y = v; vv.z = v; vv.w = v;
  for (unsigned i = blockIdx.x * blockDim.x + threadIdx.x; i < n4; i += gridDim.x * blockDim.x)
    ((uint4*)p)[i] = vv;
}

__global__ void k_pack(const float* __restrict__ w, unsigned short* __restrict__ out, int n) {
  for (int i = blockIdx.x * blockDim.x + threadIdx.x; i < n; i += gridDim.x * blockDim.x)
    out[i] = f2bf(w[i]);
}

// ------------------------------------------------- fc_pos: [BT,3] -> bf16 [BT,256]
__global__ void k_fcpos(const float* __restrict__ p, const float* __restrict__ W,
                        const float* __restrict__ bias, unsigned short* __restrict__ A) {
  long long i = (long long)blockIdx.x * blockDim.x + threadIdx.x;   // BT*32 threads
  if (i >= (long long)BT * 32) return;
  int pt = (int)(i >> 5);
  int n0 = ((int)i & 31) << 3;
  float p0 = p[(size_t)pt * 3 + 0], p1 = p[(size_t)pt * 3 + 1], p2 = p[(size_t)pt * 3 + 2];
  union { uint4 q; unsigned short s[8]; } o;
#pragma unroll
  for (int j = 0; j < 8; ++j) {
    int n = n0 + j;
    o.s[j] = f2bf(bias[n] + p0 * W[n] + p1 * W[256 + n] + p2 * W[512 + n]);
  }
  *(uint4*)&A[(size_t)pt * LDA + n0] = o.q;
}

// One 128->128 MLP layer over a wave-local 16x128 LDS tile (in place).
// wsrc must be a compile-time-distinct LDS pointer so 16-bit ds_loads are emitted.
__device__ __forceinline__ void planenet_layer(unsigned short* hrow,
                                               const unsigned short* wsrc,
                                               const float* __restrict__ bsrc,
                                               int lane, bool relu) {
  v8f z = {0.f,0.f,0.f,0.f,0.f,0.f,0.f,0.f};
  v8f acc[8];
#pragma unroll
  for (int nc = 0; nc < 8; ++nc) acc[nc] = z;
#pragma unroll
  for (int kt = 0; kt < 4; ++kt) {
    AFrag a = afrag_rows(hrow, 128, kt * 32, lane);
#pragma unroll
    for (int nc = 0; nc < 8; ++nc) {
      AFrag bb = bfrag(wsrc, kt, nc, lane);
      acc[nc] = wmma_bf16(a, bb, acc[nc]);
    }
  }
  int nlo = lane & 15, mhi = (lane & 16) ? 8 : 0;
#pragma unroll
  for (int nc = 0; nc < 8; ++nc) {
    int n = nc * 16 + nlo;
    float bi = bsrc[n];
#pragma unroll
    for (int r = 0; r < 8; ++r) {
      float v = acc[nc][r] + bi;
      if (relu && v < 0.f) v = 0.f;
      hrow[(r + mhi) * 128 + n] = f2bf(v);
    }
  }
}

// ------------------------------------------------- FCPlanenet + global max pool
__global__ void k_planenet(const float* __restrict__ p,
                           const float* __restrict__ W0, const float* __restrict__ b0,
                           const unsigned short* __restrict__ pw1, const float* __restrict__ b1,
                           const unsigned short* __restrict__ pw2, const float* __restrict__ b2,
                           const unsigned short* __restrict__ pw3, const float* __restrict__ b3,
                           unsigned* __restrict__ enc_out) {
  extern __shared__ char smem[];
  unsigned short* s_h  = (unsigned short*)smem;            // 128x128
  unsigned short* s_w1 = s_h  + 16384;                     // 128x128 bf16
  unsigned short* s_w2 = s_w1 + 16384;
  unsigned short* s_w3 = s_w2 + 16384;

  int b = blockIdx.x / NT, tile = blockIdx.x % NT;
  // stage packed weights (L2 resident) into LDS
  for (int i = threadIdx.x; i < 2048; i += blockDim.x) {
    ((uint4*)s_w1)[i] = ((const uint4*)pw1)[i];
    ((uint4*)s_w2)[i] = ((const uint4*)pw2)[i];
    ((uint4*)s_w3)[i] = ((const uint4*)pw3)[i];
  }
  // h0 = p @ W0 + b0   (K=3, scalar)
  for (int i = threadIdx.x; i < 16384; i += blockDim.x) {
    int m = i >> 7, n = i & 127;
    int t = tile * 128 + m; if (t >= TPT) t = TPT - 1;
    const float* pp = p + ((size_t)b * TPT + t) * 3;
    s_h[i] = f2bf(b0[n] + pp[0] * W0[n] + pp[1] * W0[128 + n] + pp[2] * W0[256 + n]);
  }
  __syncthreads();

  int w = threadIdx.x >> 5, lane = threadIdx.x & 31;
  unsigned short* hrow = s_h + (w * 16) * 128;

  // manually unrolled layers: each call sees a distinct constant LDS base,
  // so B-frag reads lower to 16-bit ds_load (not flat_load via pointer select)
  planenet_layer(hrow, s_w1, b1, lane, true);
  planenet_layer(hrow, s_w2, b2, lane, true);
  planenet_layer(hrow, s_w3, b3, lane, false);

  __syncthreads();
  if (threadIdx.x < 128) {
    int n = threadIdx.x;
    float mx = -3.4e38f;
    for (int m = 0; m < 128; ++m) {
      int t = tile * 128 + m;
      if (t < TPT) mx = fmaxf(mx, bf2f(s_h[m * 128 + n]));
    }
    atomicMax(&enc_out[b * 128 + n], encf(mx));
  }
}

// ------------------------------------------------- normals, basis, hdim
__global__ void k_basis(const unsigned* __restrict__ enc,
                        const float* __restrict__ ppW0, const float* __restrict__ ppb0,
                        const float* __restrict__ ppW1, const float* __restrict__ ppb1,
                        const float* __restrict__ hdW0, const float* __restrict__ hdb0,
                        const float* __restrict__ hdW1, const float* __restrict__ hdb1,
                        float* __restrict__ nhdim, float* __restrict__ cmat,
                        float* __restrict__ out_tail) {
  __shared__ float r[8][128];
  __shared__ float nf[2][8][3];
  for (int i = threadIdx.x; i < 1024; i += blockDim.x) {
    unsigned e = enc[i];
    float v = (e == 0u) ? 0.f : decf(e);
    r[i >> 7][i & 127] = fmaxf(v, 0.f);               // relu(net_pl)
  }
  __syncthreads();
  if (threadIdx.x < 48) {
    int task = threadIdx.x;
    int l = task / 24, b = (task % 24) / 3, j = task % 3;
    const float* W = l ? ppW1 : ppW0;
    const float* bi = l ? ppb1 : ppb0;
    float acc = bi[j];
    for (int n = 0; n < 128; ++n) acc += r[b][n] * W[n * 3 + j];
    nf[l][b][j] = acc;
  }
  __syncthreads();
  for (int i = threadIdx.x; i < 2048; i += blockDim.x) {
    int l = i >> 10, b = (i >> 7) & 7, n = i & 127;
    const float* W = l ? hdW1 : hdW0;
    const float* bi = l ? hdb1 : hdb0;
    nhdim[i] = bi[n] + nf[l][b][0] * W[n] + nf[l][b][1] * W[128 + n] + nf[l][b][2] * W[256 + n];
  }
  if (threadIdx.x < 16) {
    int b = threadIdx.x >> 1, l = threadIdx.x & 1;
    float nx = nf[l][b][0], ny = nf[l][b][1], nz = nf[l][b][2];
    float il = rsqrtf(nx * nx + ny * ny + nz * nz + 1e-30f);
    nx *= il; ny *= il; nz *= il;
    float vx = -ny, vy = nx, vz = 0.f;                // cross(bz, n)
    float wx = 0.f, wy = -nz, wz = ny;                // cross(bx, n)
    if (vx == 0.f) vx = wx;
    if (vy == 0.f) vy = wy;
    if (vz == 0.f) vz = wz;
    float ux = ny * vz - nz * vy, uy = nz * vx - nx * vz, uz = nx * vy - ny * vx;
    float ivl = rsqrtf(vx * vx + vy * vy + vz * vz + 1e-30f);
    float iul = rsqrtf(ux * ux + uy * uy + uz * uz + 1e-30f);
    vx *= ivl; vy *= ivl; vz *= ivl;
    ux *= iul; uy *= iul; uz *= iul;
    float rows[9] = { vx, vy, vz, ux, uy, uz, nx, ny, nz };
    int base = (b * 2 + l) * 9;
    for (int k = 0; k < 9; ++k) { cmat[base + k] = rows[k]; out_tail[base + k] = rows[k]; }
  }
}

// ------------------------------------------------- per-plane bin indices
__global__ void k_idx(const float* __restrict__ p, const float* __restrict__ cmat,
                      int* __restrict__ idxs) {
  int i = blockIdx.x * blockDim.x + threadIdx.x;
  if (i >= BT) return;
  int b = i / TPT;
  float p0 = p[(size_t)i * 3 + 0], p1 = p[(size_t)i * 3 + 1], p2 = p[(size_t)i * 3 + 2];
  const float INV = 1.0f / 1.101f;                    // 1/(1+PAD+1e-3)
  auto grid1 = [&](float x) -> int {
    float nx = x * INV + 0.5f;
    nx = fminf(fmaxf(nx, 0.0f), 1.0f - 1e-5f);
    return (int)(nx * 128.0f);
  };
  // static planes: xz, xy, yz
  idxs[0 * BT + i] = grid1(p0) + 128 * grid1(p2);
  idxs[1 * BT + i] = grid1(p0) + 128 * grid1(p1);
  idxs[2 * BT + i] = grid1(p1) + 128 * grid1(p2);
#pragma unroll
  for (int l = 0; l < 2; ++l) {
    const float* C = cmat + (b * 2 + l) * 9;
    float q0 = p0 * C[0] + p1 * C[1] + p2 * C[2];
    float q1 = p0 * C[3] + p1 * C[4] + p2 * C[5];
    idxs[(3 + l) * BT + i] = grid1(q0) + 128 * grid1(q1);
  }
}

// ------------------------------------------------- fused resblock (3 GEMMs, WMMA)
__global__ void k_resblock(unsigned short* __restrict__ A,
                           const unsigned short* __restrict__ pw0,
                           const unsigned short* __restrict__ pw1,
                           const unsigned short* __restrict__ pwsc,
                           const float* __restrict__ b0, const float* __restrict__ b1) {
  extern __shared__ char smem[];
  unsigned short* s_w0  = (unsigned short*)smem;        // 256x128 bf16 = 64KB
  unsigned short* s_wsc = s_w0 + 32768;                 // 256x128 bf16 = 64KB
  unsigned short* s_w1  = s_wsc + 32768;                // 128x128 bf16 = 32KB
  unsigned short* s_h   = s_w1 + 16384;                 // 128x128 bf16 = 32KB
  for (int i = threadIdx.x; i < 4096; i += blockDim.x) {
    ((uint4*)s_w0)[i]  = ((const uint4*)pw0)[i];
    ((uint4*)s_wsc)[i] = ((const uint4*)pwsc)[i];
    if (i < 2048) ((uint4*)s_w1)[i] = ((const uint4*)pw1)[i];
  }
  __syncthreads();

  int w = threadIdx.x >> 5, lane = threadIdx.x & 31;
  int b = blockIdx.x / NT, tile = blockIdx.x % NT;
  int t0 = tile * 128 + w * 16;
  unsigned short* Ab = A + (size_t)b * TPT * LDA;
  int nlo = lane & 15, mhi = (lane & 16) ? 8 : 0;
  v8f z = {0.f,0.f,0.f,0.f,0.f,0.f,0.f,0.f};

  // netb = relu(x) @ W0  (K=256)
  v8f accH[8];
#pragma unroll
  for (int nc = 0; nc < 8; ++nc) accH[nc] = z;
#pragma unroll
  for (int kt = 0; kt < 8; ++kt) {
    AFrag a = afrag_gclamp(Ab, LDA, kt * 32, t0, TPT - 1, lane, 1);
#pragma unroll
    for (int nc = 0; nc < 8; ++nc) {
      AFrag bb = bfrag(s_w0, kt, nc, lane);
      accH[nc] = wmma_bf16(a, bb, accH[nc]);
    }
  }
  // hb = relu(netb + b0) staged to LDS (wave-local 16x128 tile)
#pragma unroll
  for (int nc = 0; nc < 8; ++nc) {
    int n = nc * 16 + nlo;
    float bi = b0[n];
#pragma unroll
    for (int r = 0; r < 8; ++r) {
      float v = accH[nc][r] + bi;
      if (v < 0.f) v = 0.f;
      s_h[(w * 16 + r + mhi) * 128 + n] = f2bf(v);
    }
  }
  // dx = hb @ W1 (K=128), then accumulate sc = x @ Wsc (K=256)
  v8f acc[8];
#pragma unroll
  for (int nc = 0; nc < 8; ++nc) acc[nc] = z;
#pragma unroll
  for (int kt = 0; kt < 4; ++kt) {
    AFrag a = afrag_rows(s_h + (w * 16) * 128, 128, kt * 32, lane);
#pragma unroll
    for (int nc = 0; nc < 8; ++nc) {
      AFrag bb = bfrag(s_w1, kt, nc, lane);
      acc[nc] = wmma_bf16(a, bb, acc[nc]);
    }
  }
#pragma unroll
  for (int kt = 0; kt < 8; ++kt) {
    AFrag a = afrag_gclamp(Ab, LDA, kt * 32, t0, TPT - 1, lane, 0);
#pragma unroll
    for (int nc = 0; nc < 8; ++nc) {
      AFrag bb = bfrag(s_wsc, kt, nc, lane);
      acc[nc] = wmma_bf16(a, bb, acc[nc]);
    }
  }
  // out = sc + dx + b1  -> A[:, 0:128]
#pragma unroll
  for (int nc = 0; nc < 8; ++nc) {
    int n = nc * 16 + nlo;
    float bi = b1[n];
#pragma unroll
    for (int r = 0; r < 8; ++r) {
      int t = t0 + r + mhi;
      if (t < TPT) Ab[(size_t)t * LDA + n] = f2bf(acc[nc][r] + bi);
    }
  }
}

// ------------------------------------------------- scatter-max into bins
__global__ void k_smax(const unsigned short* __restrict__ A, const int* __restrict__ idx,
                       unsigned* __restrict__ bins) {
  long long i = (long long)blockIdx.x * blockDim.x + threadIdx.x;   // BT*32
  if (i >= (long long)BT * 32) return;
  int pt = (int)(i >> 5);
  int fb = ((int)i & 31) << 2;
  int b = pt / TPT;
  int id = idx[pt];
  union { uint2 d; unsigned short s[4]; } v;
  v.d = *(const uint2*)&A[(size_t)pt * LDA + fb];
  unsigned* bp = bins + ((size_t)(b * R2 + id)) * 128 + fb;
#pragma unroll
  for (int j = 0; j < 4; ++j) atomicMax(&bp[j], encf(bf2f(v.s[j])));
}

// ------------------------------------------------- gather + accumulate pooled
__global__ void k_gather(unsigned short* __restrict__ A, const int* __restrict__ idx,
                         const unsigned* __restrict__ bins, int accumulate) {
  long long i = (long long)blockIdx.x * blockDim.x + threadIdx.x;   // BT*32
  if (i >= (long long)BT * 32) return;
  int pt = (int)(i >> 5);
  int fb = ((int)i & 31) << 2;
  int b = pt / TPT;
  int id = idx[pt];
  const unsigned* bp = bins + ((size_t)(b * R2 + id)) * 128 + fb;
  unsigned short* dst = &A[(size_t)pt * LDA + 128 + fb];
  union { uint2 d; unsigned short s[4]; } o;
  if (accumulate) o.d = *(const uint2*)dst;
#pragma unroll
  for (int j = 0; j < 4; ++j) {
    unsigned e = bp[j];
    float v = (e == 0u) ? 0.f : decf(e);              // empty bin -> 0
    if (accumulate) v += bf2f(o.s[j]);
    o.s[j] = f2bf(v);
  }
  *(uint2*)dst = o.d;
}

// ------------------------------------------------- per-plane bin counts
__global__ void k_counts(const int* __restrict__ idxs, unsigned* __restrict__ counts) {
  int i = blockIdx.x * blockDim.x + threadIdx.x;      // 5*BT
  if (i >= 5 * BT) return;
  int l = i / BT, pt = i % BT, b = pt / TPT;
  int id = idxs[(size_t)l * BT + pt];
  atomicAdd(&counts[(size_t)(l * NB + b) * R2 + id], 1u);
}

// ------------------------------------------------- fc_c + scatter-mean sums (WMMA)
__global__ void k_fcc(const unsigned short* __restrict__ A,
                      const unsigned short* __restrict__ pwc, const float* __restrict__ bc,
                      const float* __restrict__ nhdim, const int* __restrict__ idxs,
                      float* __restrict__ out) {
  extern __shared__ char smem[];
  unsigned short* s_w = (unsigned short*)smem;        // 128x128 bf16
  for (int i = threadIdx.x; i < 2048; i += blockDim.x)
    ((uint4*)s_w)[i] = ((const uint4*)pwc)[i];
  __syncthreads();

  int w = threadIdx.x >> 5, lane = threadIdx.x & 31;
  int b = blockIdx.x / NT, tile = blockIdx.x % NT;
  int t0 = tile * 128 + w * 16;
  const unsigned short* Ab = A + (size_t)b * TPT * LDA;
  int nlo = lane & 15, mhi = (lane & 16) ? 8 : 0;
  v8f z = {0.f,0.f,0.f,0.f,0.f,0.f,0.f,0.f};
  v8f acc[8];
#pragma unroll
  for (int nc = 0; nc < 8; ++nc) acc[nc] = z;
#pragma unroll
  for (int kt = 0; kt < 4; ++kt) {
    AFrag a = afrag_gclamp(Ab, LDA, kt * 32, t0, TPT - 1, lane, 0);
#pragma unroll
    for (int nc = 0; nc < 8; ++nc) {
      AFrag bb = bfrag(s_w, kt, nc, lane);
      acc[nc] = wmma_bf16(a, bb, acc[nc]);
    }
  }
#pragma unroll
  for (int nc = 0; nc < 8; ++nc) {
    int n = nc * 16 + nlo;
    float bi = bc[n];
#pragma unroll
    for (int r = 0; r < 8; ++r) {
      int t = t0 + r + mhi;
      if (t >= TPT) continue;
      float cv = acc[nc][r] + bi;
      size_t pt = (size_t)b * TPT + t;
#pragma unroll
      for (int l = 0; l < 5; ++l) {
        float val = cv + ((l >= 3) ? nhdim[(size_t)((l - 3) * NB + b) * 128 + n] : 0.f);
        int id = idxs[(size_t)l * BT + pt];
        atomicAdd(&out[(((size_t)(b * 5 + l) * 128 + n) << 14) + id], val);
      }
    }
  }
}

// ------------------------------------------------- mean normalize
__global__ void k_norm(float* __restrict__ out, const unsigned* __restrict__ counts) {
  const long long total = (long long)NB * 5 * 128 * R2;
  for (long long i = (long long)blockIdx.x * blockDim.x + threadIdx.x; i < total;
       i += (long long)gridDim.x * blockDim.x) {
    int j = (int)(i & (R2 - 1));
    long long rest = i >> 14;
    int n = (int)(rest & 127); (void)n;
    long long rest2 = rest >> 7;
    int l = (int)(rest2 % 5);
    int b = (int)(rest2 / 5);
    unsigned c = counts[(size_t)(l * NB + b) * R2 + j];
    out[i] = c ? (out[i] / (float)c) : 0.0f;
  }
}

// ================================================================ host
extern "C" void kernel_launch(void* const* d_in, const int* in_sizes, int n_in,
                              void* d_out, int out_size, void* d_ws, size_t ws_size,
                              hipStream_t stream) {
  (void)in_sizes; (void)n_in; (void)out_size; (void)ws_size;
  const float* p       = (const float*)d_in[0];
  const float* fcpos_W = (const float*)d_in[1];
  const float* fcpos_b = (const float*)d_in[2];
  const float *bW0[5], *bb0[5], *bW1[5], *bb1[5], *bWsc[5];
  for (int i = 0; i < 5; ++i) {
    bW0[i]  = (const float*)d_in[3 + 5 * i];
    bb0[i]  = (const float*)d_in[4 + 5 * i];
    bW1[i]  = (const float*)d_in[5 + 5 * i];
    bb1[i]  = (const float*)d_in[6 + 5 * i];
    bWsc[i] = (const float*)d_in[7 + 5 * i];
  }
  const float* fcc_W = (const float*)d_in[28];
  const float* fcc_b = (const float*)d_in[29];
  const float *plW[4], *plb[4];
  for (int i = 0; i < 4; ++i) { plW[i] = (const float*)d_in[30 + 2 * i]; plb[i] = (const float*)d_in[31 + 2 * i]; }
  const float* ppW[2] = { (const float*)d_in[38], (const float*)d_in[40] };
  const float* ppb[2] = { (const float*)d_in[39], (const float*)d_in[41] };
  const float* hdW[2] = { (const float*)d_in[42], (const float*)d_in[44] };
  const float* hdb[2] = { (const float*)d_in[43], (const float*)d_in[45] };

  char* ws = (char*)d_ws;
  size_t off = 0;
  auto carve = [&](size_t bytes) -> char* {
    char* r = ws + off;
    off = (off + bytes + 255) & ~(size_t)255;
    return r;
  };
  unsigned short* Abuf = (unsigned short*)carve((size_t)BT * LDA * 2);   // 122.9 MB bf16 acts
  unsigned*       bins = (unsigned*)carve((size_t)NB * R2 * 128 * 4);    // 64 MB (L2-resident)
  int*            idxs = (int*)carve((size_t)5 * BT * 4);
  unsigned*     counts = (unsigned*)carve((size_t)5 * NB * R2 * 4);
  unsigned*      netpl = (unsigned*)carve(1024 * 4);
  float*          cmat = (float*)carve(144 * 4);
  float*         nhdim = (float*)carve(2048 * 4);
  unsigned short* pk_pl[3];
  for (int i = 0; i < 3; ++i) pk_pl[i] = (unsigned short*)carve(16384 * 2);
  unsigned short *pk_w0[5], *pk_w1[5], *pk_wsc[5];
  for (int i = 0; i < 5; ++i) {
    pk_w0[i]  = (unsigned short*)carve(32768 * 2);
    pk_w1[i]  = (unsigned short*)carve(16384 * 2);
    pk_wsc[i] = (unsigned short*)carve(32768 * 2);
  }
  unsigned short* pk_fcc = (unsigned short*)carve(16384 * 2);

  float* out = (float*)d_out;
  const size_t PLANES_ELEMS = (size_t)NB * 5 * 128 * R2;   // 83,886,080

  // d_out accumulates atomics -> must be zeroed every call (graph-replay safe)
  k_fill<<<8192, 256, 0, stream>>>((unsigned*)out, 0u, (unsigned)PLANES_ELEMS);
  k_fill<<<64, 256, 0, stream>>>(counts, 0u, 5u * NB * R2);
  k_fill<<<1, 256, 0, stream>>>(netpl, 0u, 1024u);

  // pre-convert all WMMA weights to bf16
  for (int i = 0; i < 3; ++i) k_pack<<<64, 256, 0, stream>>>(plW[i + 1], pk_pl[i], 16384);
  for (int i = 0; i < 5; ++i) {
    k_pack<<<128, 256, 0, stream>>>(bW0[i],  pk_w0[i],  32768);
    k_pack<<<64,  256, 0, stream>>>(bW1[i],  pk_w1[i],  16384);
    k_pack<<<128, 256, 0, stream>>>(bWsc[i], pk_wsc[i], 32768);
  }
  k_pack<<<64, 256, 0, stream>>>(fcc_W, pk_fcc, 16384);

  // fc_pos -> bf16 activations
  k_fcpos<<<BT * 32 / 256, 256, 0, stream>>>(p, fcpos_W, fcpos_b, Abuf);

  // FCPlanenet (WMMA) + max pool, then basis / hdim / indices
  k_planenet<<<NB * NT, 256, 131072, stream>>>(p, plW[0], plb[0],
                                               pk_pl[0], plb[1], pk_pl[1], plb[2], pk_pl[2], plb[3],
                                               netpl);
  k_basis<<<1, 64, 0, stream>>>(netpl, ppW[0], ppb[0], ppW[1], ppb[1],
                                hdW[0], hdb[0], hdW[1], hdb[1],
                                nhdim, cmat, out + PLANES_ELEMS);
  k_idx<<<(BT + 255) / 256, 256, 0, stream>>>(p, cmat, idxs);

  // resblock 0 (no pooling input)
  k_resblock<<<NB * NT, 256, 196608, stream>>>(Abuf, pk_w0[0], pk_w1[0], pk_wsc[0], bb0[0], bb1[0]);

  // blocks 1..4: 5-plane scatter-max pooling (one L2-resident bin buffer reused) + resblock
  for (int s = 1; s < 5; ++s) {
    for (int l = 0; l < 5; ++l) {
      k_fill<<<8192, 256, 0, stream>>>(bins, 0u, (unsigned)(NB * R2 * 128));
      k_smax<<<BT * 32 / 256, 256, 0, stream>>>(Abuf, idxs + (size_t)l * BT, bins);
      k_gather<<<BT * 32 / 256, 256, 0, stream>>>(Abuf, idxs + (size_t)l * BT, bins, l > 0 ? 1 : 0);
    }
    k_resblock<<<NB * NT, 256, 196608, stream>>>(Abuf, pk_w0[s], pk_w1[s], pk_wsc[s], bb0[s], bb1[s]);
  }

  // fc_c (WMMA) fused with scatter-mean accumulation, then normalize
  k_counts<<<(5 * BT + 255) / 256, 256, 0, stream>>>(idxs, counts);
  k_fcc<<<NB * NT, 256, 32768, stream>>>(Abuf, pk_fcc, fcc_b, nhdim, idxs, out);
  k_norm<<<65536, 256, 0, stream>>>(out, counts);
}
